// MOELayer_23304492548266
// MI455X (gfx1250) — compile-verified, matrix operand order checked by reading, not run on hardware
//
#include <hip/hip_runtime.h>

// ---------------- problem constants ----------------
#define S_TOK 16384
#define MDIM  1024
#define HDIM  2048
#define NEXP  16
#define CAP   1024   // capacity per expert; note NEXP*CAP == S_TOK

// ---------------- vector typedefs ----------------
typedef __attribute__((ext_vector_type(8)))  short   sv8;    // 16B of bf16 bits
typedef __attribute__((ext_vector_type(16))) short   sv16;
typedef __attribute__((ext_vector_type(16))) __bf16  bf16x16;
typedef __attribute__((ext_vector_type(8)))  float   f32x8;
typedef __attribute__((ext_vector_type(4)))  short   sv4;

// float -> bf16 bits, round-to-nearest-even
__device__ __forceinline__ unsigned short f2bf(float f) {
    unsigned u = __float_as_uint(f);
    u += 0x7FFFu + ((u >> 16) & 1u);
    return (unsigned short)(u >> 16);
}

// Load one 16x32 bf16 WMMA operand fragment for this lane from an LDS row.
__device__ __forceinline__ bf16x16 ldfrag(const unsigned short* rowp, int kb) {
    sv8 lo = *(const sv8*)(rowp + kb);
    sv8 hi = *(const sv8*)(rowp + kb + 16);
    sv16 s = __builtin_shufflevector(lo, hi, 0,1,2,3,4,5,6,7,8,9,10,11,12,13,14,15);
    return __builtin_bit_cast(bf16x16, s);
}

// ---------------- GEMM config ----------------
#define BK   32
#define LDST 56   // LDS row stride (shorts): 112B, 16B aligned, conflict-friendly

// Async copy of one 128xBK bf16 A tile: global -> LDS, b128 per lane,
// tracked by ASYNCcnt (CDNA5 async-tensor path; no VGPR staging).
__device__ __forceinline__ void tileA_async(
    unsigned ldsA, const unsigned short* gbase, int ldg, int kk, int tid)
{
#pragma unroll
    for (int i = 0; i < 2; ++i) {
        int idx = tid + i * 256;
        int r = idx >> 2, kc = (idx & 3) * 8;
        unsigned dst = ldsA + (unsigned)((r * LDST + kc) * 2);
        unsigned long long src =
            (unsigned long long)(uintptr_t)(gbase + (size_t)r * ldg + kk + kc);
        asm volatile("global_load_async_to_lds_b128 %0, %1, off"
                     :: "v"(dst), "v"(src) : "memory");
    }
}

__device__ __forceinline__ void asyncs_done() {
    asm volatile("s_wait_asynccnt 0" ::: "memory");
}

// B tile: 4 coalesced b32 loads per group into registers (issued before compute)
__device__ __forceinline__ void tileB_ld(
    float r[16], const float* Bbase, int ldb, int kk, int tid)
{
#pragma unroll
    for (int g = 0; g < 4; ++g) {
        int gid = tid + g * 256;
        int n = gid & 127, k0 = (gid >> 7) * 4;
#pragma unroll
        for (int j = 0; j < 4; ++j)
            r[g * 4 + j] = Bbase[(size_t)(kk + k0 + j) * ldb + n];
    }
}

// ... converted + stored transposed [n][k] as packed 8B DS stores (after compute)
__device__ __forceinline__ void tileB_st(
    unsigned short* lB, const float r[16], int tid)
{
#pragma unroll
    for (int g = 0; g < 4; ++g) {
        int gid = tid + g * 256;
        int n = gid & 127, k0 = (gid >> 7) * 4;
        sv4 o;
        o[0] = (short)f2bf(r[g * 4 + 0]);
        o[1] = (short)f2bf(r[g * 4 + 1]);
        o[2] = (short)f2bf(r[g * 4 + 2]);
        o[3] = (short)f2bf(r[g * 4 + 3]);
        *(sv4*)&lB[n * LDST + k0] = o;
    }
}

__device__ __forceinline__ void tile_mma(
    f32x8 acc[2][4], const unsigned short* lA, const unsigned short* lB,
    int wm, int wn, int lane)
{
    const int kb = (lane >> 4) * 8;
    bf16x16 af[2], bfg[4];
#pragma unroll
    for (int tm = 0; tm < 2; ++tm)
        af[tm] = ldfrag(&lA[(wm * 32 + tm * 16 + (lane & 15)) * LDST], kb);
#pragma unroll
    for (int tn = 0; tn < 4; ++tn)
        bfg[tn] = ldfrag(&lB[(wn * 64 + tn * 16 + (lane & 15)) * LDST], kb);
#pragma unroll
    for (int tm = 0; tm < 2; ++tm)
#pragma unroll
        for (int tn = 0; tn < 4; ++tn)
            acc[tm][tn] = __builtin_amdgcn_wmma_f32_16x16x32_bf16(
                false, af[tm], false, bfg[tn], (short)0, acc[tm][tn],
                false, false);
}

// ---------------- kernel 0: zero me/ce ----------------
__global__ void moe_init(float* me, float* ce) {
    int t = threadIdx.x;
    if (t < NEXP) { me[t] = 0.f; ce[t] = 0.f; }
}

// ---------------- kernel 1: gating (one wave per token) ----------------
__global__ __launch_bounds__(256) void moe_gate(
    const float* __restrict__ x, const float* __restrict__ wg,
    int* __restrict__ eidx, float* __restrict__ gate,
    int* __restrict__ tos, float* __restrict__ me, float* __restrict__ ce)
{
    __shared__ float sme[NEXP];
    __shared__ float sce[NEXP];
    const int tid  = threadIdx.x;
    const int lane = tid & 31;
    const int wid  = tid >> 5;
    const int t    = blockIdx.x * 8 + wid;

    if (tid < NEXP) { sme[tid] = 0.f; sce[tid] = 0.f; }
    __syncthreads();

    if (lane == 0) tos[t] = -1;   // reset inverse map (S_TOK == NEXP*CAP)

    float lg[NEXP];
#pragma unroll
    for (int e = 0; e < NEXP; ++e) lg[e] = 0.f;

    const float4* xr = (const float4*)(x + (size_t)t * MDIM);
#pragma unroll
    for (int j = 0; j < 8; ++j) {
        float4 xv = xr[j * 32 + lane];
#pragma unroll
        for (int e = 0; e < NEXP; ++e) {
            float4 wv = ((const float4*)(wg + (size_t)e * MDIM))[j * 32 + lane];
            lg[e] += xv.x * wv.x + xv.y * wv.y + xv.z * wv.z + xv.w * wv.w;
        }
    }
#pragma unroll
    for (int e = 0; e < NEXP; ++e) {
#pragma unroll
        for (int off = 16; off > 0; off >>= 1)
            lg[e] += __shfl_xor(lg[e], off, 32);
    }

    float mx = lg[0]; int bi = 0;
#pragma unroll
    for (int e = 1; e < NEXP; ++e)
        if (lg[e] > mx) { mx = lg[e]; bi = e; }   // first-max like argmax
    float ssum = 0.f;
#pragma unroll
    for (int e = 0; e < NEXP; ++e) ssum = ssum + __expf(lg[e] - mx);
    float g = 1.0f / ssum;

    if (lane == 0) {
        eidx[t] = bi;
        gate[t] = g;
#pragma unroll
        for (int e = 0; e < NEXP; ++e)
            atomicAdd(&sme[e], __expf(lg[e] - mx) / ssum);
        atomicAdd(&sce[bi], 1.0f);
    }
    __syncthreads();
    if (tid < NEXP) {
        atomicAdd(&me[tid], sme[tid]);
        atomicAdd(&ce[tid], sce[tid]);
    }
}

// ---------------- kernel 2: in-order capacity assignment + l_aux ----------
__global__ void moe_order(const int* __restrict__ eidx, int* __restrict__ slotv,
                          const float* __restrict__ me, const float* __restrict__ ce,
                          float* __restrict__ laux_out)
{
    const int lane = threadIdx.x;
    const unsigned lt = (1u << lane) - 1u;
    int cnt[NEXP];
#pragma unroll
    for (int e = 0; e < NEXP; ++e) cnt[e] = 0;

    for (int it = 0; it < S_TOK / 32; ++it) {
        const int t = it * 32 + lane;
        const int e = eidx[t];
        int loc = 0;
#pragma unroll
        for (int ex = 0; ex < NEXP; ++ex) {
            unsigned m = (unsigned)__ballot(e == ex);
            if (e == ex) loc = cnt[ex] + __popc(m & lt);
            cnt[ex] += __popc(m);
        }
        slotv[t] = (loc < CAP) ? (e * CAP + loc) : -1;
    }

    float p = (lane < NEXP) ? me[lane] * ce[lane] : 0.f;
#pragma unroll
    for (int off = 16; off > 0; off >>= 1) p += __shfl_xor(p, off, 32);
    if (lane == 0)
        *laux_out = p * ((float)NEXP / ((float)S_TOK * (float)S_TOK));
}

// ---------------- kernel 3: dispatch (wave per token) ----------------
__global__ __launch_bounds__(256) void moe_dispatch(
    const float* __restrict__ x, const int* __restrict__ slotv,
    unsigned short* __restrict__ disp, int* __restrict__ tos,
    float* __restrict__ y)
{
    const int tid  = threadIdx.x;
    const int lane = tid & 31;
    const int t    = blockIdx.x * 8 + (tid >> 5);
    const int sl   = slotv[t];

    if (sl >= 0) {
        const float4* xr = (const float4*)(x + (size_t)t * MDIM);
        unsigned short* dr = disp + (size_t)sl * MDIM;
#pragma unroll
        for (int j = 0; j < 8; ++j) {
            float4 v = xr[j * 32 + lane];
            sv4 o;
            o[0] = (short)f2bf(v.x); o[1] = (short)f2bf(v.y);
            o[2] = (short)f2bf(v.z); o[3] = (short)f2bf(v.w);
            *(sv4*)(dr + (size_t)(j * 32 + lane) * 4) = o;
        }
        if (lane == 0) tos[sl] = t;
    } else {
        float4 z{0.f, 0.f, 0.f, 0.f};
        float4* yr = (float4*)(y + (size_t)t * MDIM);
#pragma unroll
        for (int j = 0; j < 8; ++j) yr[j * 32 + lane] = z;
    }
}

// ---------------- kernel 4: h = relu(disp @ w1 + b1), bf16 out -----------
__global__ __launch_bounds__(256) void moe_gemm1(
    const unsigned short* __restrict__ disp, const float* __restrict__ w1,
    const float* __restrict__ b1, unsigned short* __restrict__ hbuf)
{
    __shared__ alignas(16) unsigned short lA[2][128 * LDST];
    __shared__ alignas(16) unsigned short lB[2][128 * LDST];
    const int e  = blockIdx.z;
    const int n0 = blockIdx.x * 128;   // over HDIM
    const int c0 = blockIdx.y * 128;   // over CAP
    const int tid = threadIdx.x, lane = tid & 31, wid = tid >> 5;
    const int wm = wid & 3, wn = wid >> 2;

    f32x8 acc[2][4];
#pragma unroll
    for (int i = 0; i < 2; ++i)
#pragma unroll
        for (int j = 0; j < 4; ++j)
            acc[i][j] = f32x8{0.f,0.f,0.f,0.f,0.f,0.f,0.f,0.f};

    const unsigned short* Abase = disp + ((size_t)e * CAP + c0) * MDIM;
    const float* Bbase = w1 + (size_t)e * MDIM * HDIM + n0;
    const unsigned ldsA0 = (unsigned)(uintptr_t)&lA[0][0];
    const unsigned ldsA1 = (unsigned)(uintptr_t)&lA[1][0];

    // prologue: stage 0
    float br[16];
    tileA_async(ldsA0, Abase, MDIM, 0, tid);
    tileB_ld(br, Bbase, HDIM, 0, tid);
    tileB_st(lB[0], br, tid);
    asyncs_done();
    __syncthreads();

    const int nit = MDIM / BK;
    for (int it = 0; it < nit; ++it) {
        const int cur = it & 1, nxt = cur ^ 1;
        const int knxt = (it + 1) * BK;
        if (it + 1 < nit) {
            tileA_async(nxt ? ldsA1 : ldsA0, Abase, MDIM, knxt, tid);
            tileB_ld(br, Bbase, HDIM, knxt, tid);
        }
        tile_mma(acc, lA[cur], lB[cur], wm, wn, lane);
        if (it + 1 < nit) tileB_st(lB[nxt], br, tid);
        asyncs_done();
        __syncthreads();
    }

    // epilogue: +b1, relu, bf16 store
#pragma unroll
    for (int tm = 0; tm < 2; ++tm)
#pragma unroll
        for (int tn = 0; tn < 4; ++tn) {
            const int col = n0 + wn * 64 + tn * 16 + (lane & 15);
            const float bv = b1[e * HDIM + col];
#pragma unroll
            for (int r = 0; r < 8; ++r) {
                int row = c0 + wm * 32 + tm * 16 + r + ((lane >> 4) << 3);
                float v = acc[tm][tn][r] + bv;
                v = v > 0.f ? v : 0.f;
                hbuf[((size_t)e * CAP + row) * HDIM + col] = f2bf(v);
            }
        }
}

// ---------------- kernel 5: y[token] = gate * (h @ w2 + b2) --------------
__global__ __launch_bounds__(256) void moe_gemm2(
    const unsigned short* __restrict__ hbuf, const float* __restrict__ w2,
    const float* __restrict__ b2, const int* __restrict__ tos,
    const float* __restrict__ gate, float* __restrict__ y)
{
    __shared__ alignas(16) unsigned short lA[2][128 * LDST];
    __shared__ alignas(16) unsigned short lB[2][128 * LDST];
    const int e  = blockIdx.z;
    const int n0 = blockIdx.x * 128;   // over MDIM
    const int c0 = blockIdx.y * 128;   // over CAP
    const int tid = threadIdx.x, lane = tid & 31, wid = tid >> 5;
    const int wm = wid & 3, wn = wid >> 2;

    f32x8 acc[2][4];
#pragma unroll
    for (int i = 0; i < 2; ++i)
#pragma unroll
        for (int j = 0; j < 4; ++j)
            acc[i][j] = f32x8{0.f,0.f,0.f,0.f,0.f,0.f,0.f,0.f};

    const unsigned short* Abase = hbuf + ((size_t)e * CAP + c0) * HDIM;
    const float* Bbase = w2 + (size_t)e * HDIM * MDIM + n0;
    const unsigned ldsA0 = (unsigned)(uintptr_t)&lA[0][0];
    const unsigned ldsA1 = (unsigned)(uintptr_t)&lA[1][0];

    float br[16];
    tileA_async(ldsA0, Abase, HDIM, 0, tid);
    tileB_ld(br, Bbase, MDIM, 0, tid);
    tileB_st(lB[0], br, tid);
    asyncs_done();
    __syncthreads();

    const int nit = HDIM / BK;
    for (int it = 0; it < nit; ++it) {
        const int cur = it & 1, nxt = cur ^ 1;
        const int knxt = (it + 1) * BK;
        if (it + 1 < nit) {
            tileA_async(nxt ? ldsA1 : ldsA0, Abase, HDIM, knxt, tid);
            tileB_ld(br, Bbase, MDIM, knxt, tid);
        }
        tile_mma(acc, lA[cur], lB[cur], wm, wn, lane);
        if (it + 1 < nit) tileB_st(lB[nxt], br, tid);
        asyncs_done();
        __syncthreads();
    }

    // epilogue: +b2, gate-scale, scatter directly to y[token]
#pragma unroll
    for (int tm = 0; tm < 2; ++tm)
#pragma unroll
        for (int tn = 0; tn < 4; ++tn) {
            const int col = n0 + wn * 64 + tn * 16 + (lane & 15);
            const float bv = b2[e * MDIM + col];
#pragma unroll
            for (int r = 0; r < 8; ++r) {
                int row = c0 + wm * 32 + tm * 16 + r + ((lane >> 4) << 3);
                int slot = e * CAP + row;
                int t = tos[slot];
                if (t >= 0)
                    y[(size_t)t * MDIM + col] = gate[t] * (acc[tm][tn][r] + bv);
            }
        }
}

// ---------------- host-side launcher ----------------
extern "C" void kernel_launch(void* const* d_in, const int* in_sizes, int n_in,
                              void* d_out, int out_size, void* d_ws, size_t ws_size,
                              hipStream_t stream) {
    const float* x  = (const float*)d_in[0];
    const float* wg = (const float*)d_in[1];
    const float* w1 = (const float*)d_in[2];
    const float* b1 = (const float*)d_in[3];
    const float* w2 = (const float*)d_in[4];
    const float* b2 = (const float*)d_in[5];
    float* y = (float*)d_out;                       // [S, M] + 1 (l_aux)

    char* ws = (char*)d_ws;
    unsigned short* disp = (unsigned short*)ws;             ws += (size_t)S_TOK * MDIM * 2;   // 32 MB
    unsigned short* hbuf = (unsigned short*)ws;             ws += (size_t)S_TOK * HDIM * 2;   // 64 MB
    float* gate = (float*)ws;                               ws += (size_t)S_TOK * 4;
    int*   eidx = (int*)ws;                                 ws += (size_t)S_TOK * 4;
    int*   slotv= (int*)ws;                                 ws += (size_t)S_TOK * 4;
    int*   tos  = (int*)ws;                                 ws += (size_t)S_TOK * 4;
    float* me   = (float*)ws;                               ws += 64;
    float* ce   = (float*)ws;                               ws += 64;

    moe_init<<<1, 32, 0, stream>>>(me, ce);
    moe_gate<<<S_TOK / 8, 256, 0, stream>>>(x, wg, eidx, gate, tos, me, ce);
    moe_order<<<1, 32, 0, stream>>>(eidx, slotv, me, ce, y + (size_t)S_TOK * MDIM);
    moe_dispatch<<<S_TOK / 8, 256, 0, stream>>>(x, slotv, disp, tos, y);
    moe_gemm1<<<dim3(HDIM / 128, CAP / 128, NEXP), 256, 0, stream>>>(disp, w1, b1, hbuf);
    moe_gemm2<<<dim3(MDIM / 128, CAP / 128, NEXP), 256, 0, stream>>>(hbuf, w2, b2, tos, gate, y);
}
